// ClusAttention2d_13958643712728
// MI455X (gfx1250) — matrix operand order repaired and checked by reading, not compiled, not run on hardware
//
#include <hip/hip_runtime.h>
#include <hip/hip_bf16.h>

// ---------------------------------------------------------------------------
// ClusAttention2d for MI455X (gfx1250, wave32, WMMA 16x16x32 f16->f32).
// Shapes: b=8, nw=256, nn=64, d=128, G=4 groups, OUT_CL=8 clusters.
// Workspace requirement: ~139 MB (stats + f16 weights + f_m + f_v + cl_mm).
// ---------------------------------------------------------------------------

typedef __attribute__((ext_vector_type(16))) _Float16 v16h;
typedef __attribute__((ext_vector_type(8)))  _Float16 v8h;
typedef __attribute__((ext_vector_type(8)))  float    v8f;

constexpr int B_ = 8;
constexpr int NW = 256;
constexpr int NN = 64;
constexpr int D  = 128;
constexpr int Gn = 4;
constexpr int DG = 32;      // D / Gn
constexpr int CL = 8;       // OUT_CL
constexpr float EPS_GN = 0.001f;
constexpr float PI_F = 3.14159265358979323846f;

// ---- output layout (floats), reference tuple order -------------------------
constexpr long long OFF_CLF    = 0;                                     // (b,nw,8,128)
constexpr long long OFF_CLM    = OFF_CLF    + (long long)B_*NW*CL*D;    // (b,nw,8,64)
constexpr long long OFF_CLGEO  = OFF_CLM    + (long long)B_*NW*CL*NN;   // (b,nw,8,6)
constexpr long long OFF_CLR    = OFF_CLGEO  + (long long)B_*NW*CL*6;    // (b,nw,7,1)
constexpr long long OFF_CLCOMP = OFF_CLR    + (long long)B_*NW*7;       // (b,nw,8,1)
constexpr long long OFF_AF     = OFF_CLCOMP + (long long)B_*NW*CL;      // (b,nw,64,64)
constexpr long long OFF_SCM    = OFF_AF     + (long long)B_*NW*NN*NN;   // (b,nw,1,64)
constexpr long long OFF_Q      = OFF_SCM    + (long long)B_*NW*NN;      // (b,nw,64,128)

// ---- workspace layout (floats) ---------------------------------------------
constexpr long long WS_STATF = 0;                   // 64 floats (sum,sumsq)*(b,g)
constexpr long long WS_STATQ = 64;                  // 64 floats
constexpr long long WS_WH    = 128;                 // 5 * 16384 halves = 40960 floats
constexpr long long WS_FM    = WS_WH + 40960;
constexpr long long WS_FV    = WS_FM + (long long)B_*NW*NN*D;
constexpr long long WS_CLMM  = WS_FV + (long long)B_*NW*NN*D;

// ---------------------------------------------------------------------------
// WMMA fragment helpers (CDNA5 16x16x32 f16, wave32; A/B use this code's contiguous k order)
// A 16x32: lane l -> row m0+(l&15); halves j     = K k0+16*(l>>4)+j
// B 32x16: lane l -> col n0+(l&15); halves j     = K k0+16*(l>>4)+j
// C/D 16x16 f32: lane l -> col n0+(l&15), rows m0+8*(l>>4)+r (r=vgpr)
// ---------------------------------------------------------------------------
__device__ __forceinline__ int laneid() { return threadIdx.x & 31; }

__device__ __forceinline__ v16h frag_a_lds(const _Float16* base, int m0, int k0, int ld) {
  int l = laneid();
  const _Float16* p = base + (m0 + (l & 15)) * ld + k0 + ((l >> 4) << 4);
  v8h lo = *(const v8h*)p;
  v8h hi = *(const v8h*)(p + 8);
  v16h r;
#pragma unroll
  for (int j = 0; j < 8; ++j) { r[j] = lo[j]; r[j + 8] = hi[j]; }
  return r;
}

// B fragment from an N-major (transposed) buffer: baseT[n*ldk + k]
__device__ __forceinline__ v16h frag_b_lds(const _Float16* baseT, int n0, int k0, int ldk) {
  int l = laneid();
  const _Float16* p = baseT + (n0 + (l & 15)) * ldk + k0 + ((l >> 4) << 4);
  return *(const v16h*)p;
}

// B fragment from pre-swizzled global weight buffer (see k_cvtw)
__device__ __forceinline__ v16h frag_b_gl(const _Float16* w, int kb, int nb) {
  int l = laneid();
  return *(const v16h*)(w + (((((kb << 3) + nb) << 5) + l) << 4));
}

__device__ __forceinline__ v8f wmma_f16(v16h a, v16h b, v8f c) {
  return __builtin_amdgcn_wmma_f32_16x16x32_f16(false, a, false, b, (short)0, c,
                                                false, false);
}

__device__ __forceinline__ void store_d_gl(float* out, int ld, int m0, int n0,
                                           const v8f& acc) {
  int l = laneid();
  int n  = n0 + (l & 15);
  int mb = m0 + ((l >> 4) << 3);
#pragma unroll
  for (int r = 0; r < 8; ++r) out[(mb + r) * ld + n] = acc[r];
}

__device__ __forceinline__ void store_d_lds_h(_Float16* lds, int ld, int m0, int n0,
                                              const v8f& acc) {
  int l = laneid();
  int n  = n0 + (l & 15);
  int mb = m0 + ((l >> 4) << 3);
#pragma unroll
  for (int r = 0; r < 8; ++r) lds[(mb + r) * ld + n] = (_Float16)acc[r];
}

// ---------------------------------------------------------------------------
// Small utility kernels
// ---------------------------------------------------------------------------
__global__ void k_zero(float* p) { p[threadIdx.x] = 0.0f; }  // 128 threads

// Convert 128x128 f32 weight -> fragment-swizzled f16
__global__ __launch_bounds__(256) void k_cvtw(const float* __restrict__ W,
                                              _Float16* __restrict__ out) {
  int i = blockIdx.x * 256 + threadIdx.x;
  if (i >= D * D) return;
  int k = i >> 7, n = i & 127;
  int kb = k >> 5, kr = k & 31, nb = n >> 4, nr = n & 15;
  int lane = nr + ((kr >> 4) << 4);
  int j = kr & 15;
  out[(((((kb << 3) + nb) << 5) + lane) << 4) + j] = (_Float16)W[i];
}

// Group-norm stats: each thread has a FIXED channel (stride multiple of 128),
// so it accumulates a single (sum,sumsq) pair, then LDS+global atomics.
__global__ __launch_bounds__(256) void k_stats(const float* __restrict__ x,
                                               float* __restrict__ stats) {
  constexpr int KB = 32;
  int b = blockIdx.x / KB, sub = blockIdx.x % KB;
  const float* xb = x + (long long)b * (NW * NN * D);
  int g = (((sub * 256 + threadIdx.x) & 127) >> 5);
  float s = 0.f, s2 = 0.f;
  for (long long i = sub * 256 + threadIdx.x; i < (long long)NW * NN * D;
       i += (long long)KB * 256) {
    float v = xb[i];
    s += v; s2 += v * v;
  }
  __shared__ float ls[Gn * 2];
  if (threadIdx.x < Gn * 2) ls[threadIdx.x] = 0.f;
  __syncthreads();
  atomicAdd(&ls[g * 2 + 0], s);
  atomicAdd(&ls[g * 2 + 1], s2);
  __syncthreads();
  if (threadIdx.x < Gn * 2) atomicAdd(&stats[b * Gn * 2 + threadIdx.x], ls[threadIdx.x]);
}

__global__ void k_finstats(float* stats) {
  int i = threadIdx.x;
  if (i < B_ * Gn) {
    float cnt = (float)((long long)NW * NN * DG);
    float s = stats[i * 2], s2 = stats[i * 2 + 1];
    float mu = s / cnt;
    float var = s2 / cnt - mu * mu;
    stats[i * 2 + 0] = mu;
    stats[i * 2 + 1] = rsqrtf(var + EPS_GN);
  }
}

__global__ __launch_bounds__(256) void k_normq(float* __restrict__ q,
                                               const float* __restrict__ stats) {
  long long N = (long long)B_ * NW * NN * D;
  for (long long i = (long long)blockIdx.x * 256 + threadIdx.x; i < N;
       i += (long long)gridDim.x * 256) {
    int g = ((int)(i & 127)) >> 5;
    int b = (int)(i >> 21);  // NW*NN*D = 2^21
    float mu = stats[(b * Gn + g) * 2 + 0];
    float rs = stats[(b * Gn + g) * 2 + 1];
    q[i] = (q[i] - mu) * rs;
  }
}

// ---------------------------------------------------------------------------
// k_main: per window (64x128) -> q_pre, f_v, f_m via WMMA.
// 8 waves; wave w: M-block = w>>1 (0..3), N-half = w&1 (4 tiles each).
// ---------------------------------------------------------------------------
__global__ __launch_bounds__(256) void k_main(
    const float* __restrict__ in_f, const float* __restrict__ stats,
    const _Float16* __restrict__ w1h, const _Float16* __restrict__ w2h,
    const _Float16* __restrict__ wqh, const _Float16* __restrict__ wvh,
    const float* __restrict__ b1, const float* __restrict__ b2,
    float* __restrict__ fm, float* __restrict__ fv, float* __restrict__ qout) {
  __shared__ _Float16 tA[NN * D];  // normalized _f (f16)     16KB
  __shared__ _Float16 tH[NN * D];  // hidden = leaky(_f@W1+b1) 16KB
  __shared__ float mu_s[Gn], rs_s[Gn];
  int w = blockIdx.x;
  int b = w / NW;
  long long base = (long long)w * NN * D;
  if (threadIdx.x < Gn) {
    mu_s[threadIdx.x] = stats[(b * Gn + threadIdx.x) * 2 + 0];
    rs_s[threadIdx.x] = stats[(b * Gn + threadIdx.x) * 2 + 1];
  }
  __syncthreads();
  for (int i = threadIdx.x; i < NN * D; i += 256) {
    int g = (i >> 5) & 3;
    tA[i] = (_Float16)((in_f[base + i] - mu_s[g]) * rs_s[g]);
  }
  __syncthreads();

  int wave = threadIdx.x >> 5;
  int mb   = wave >> 1;
  int nsel = wave & 1;
  v8f accQ[4] = {}, accV[4] = {}, acc1[4] = {};
#pragma unroll
  for (int kb = 0; kb < 4; ++kb) {
    v16h a = frag_a_lds(tA, mb * 16, kb * 32, D);
#pragma unroll
    for (int t = 0; t < 4; ++t) {
      int nb = (nsel << 2) + t;
      accQ[t] = wmma_f16(a, frag_b_gl(wqh, kb, nb), accQ[t]);
      accV[t] = wmma_f16(a, frag_b_gl(wvh, kb, nb), accV[t]);
      acc1[t] = wmma_f16(a, frag_b_gl(w1h, kb, nb), acc1[t]);
    }
  }
#pragma unroll
  for (int t = 0; t < 4; ++t) {
    int nb = (nsel << 2) + t;
    store_d_gl(qout + base, D, mb * 16, nb * 16, accQ[t]);
    store_d_gl(fv + base,   D, mb * 16, nb * 16, accV[t]);
    float bias = b1[nb * 16 + (laneid() & 15)];
    v8f h;
#pragma unroll
    for (int r = 0; r < 8; ++r) {
      float x = acc1[t][r] + bias;
      h[r] = x > 0.f ? x : 0.01f * x;  // leaky_relu
    }
    store_d_lds_h(tH, D, mb * 16, nb * 16, h);
  }
  __syncthreads();

  v8f acc2[4] = {};
#pragma unroll
  for (int kb = 0; kb < 4; ++kb) {
    v16h a = frag_a_lds(tH, mb * 16, kb * 32, D);
#pragma unroll
    for (int t = 0; t < 4; ++t)
      acc2[t] = wmma_f16(a, frag_b_gl(w2h, kb, (nsel << 2) + t), acc2[t]);
  }
#pragma unroll
  for (int t = 0; t < 4; ++t) {
    int nb = (nsel << 2) + t;
    float bias = b2[nb * 16 + (laneid() & 15)];
    v8f o;
#pragma unroll
    for (int r = 0; r < 8; ++r) o[r] = acc2[t][r] + bias;
    store_d_gl(fm + base, D, mb * 16, nb * 16, o);
  }
}

// ---------------------------------------------------------------------------
// k_af: S = q@q^T via WMMA (B = q^T -> row-major q tile serves B fragments),
// logits = -scl*(x2[m]+x2[n]-2S), row softmax, row min-max normalize.
// ---------------------------------------------------------------------------
__global__ __launch_bounds__(256) void k_af(const float* __restrict__ q,
                                            float* __restrict__ af_out) {
  __shared__ _Float16 tQ[NN * D];  // 16KB
  __shared__ float S[NN * NN];     // 16KB (holds logits)
  __shared__ float x2[NN];
  int w = blockIdx.x;
  long long qb = (long long)w * NN * D;
  for (int i = threadIdx.x; i < NN * D; i += 256) tQ[i] = (_Float16)q[qb + i];
  __syncthreads();
  if (threadIdx.x < NN) {
    float s = 0.f;
    for (int j = 0; j < D; ++j) {
      float v = (float)tQ[threadIdx.x * D + j];
      s += v * v;
    }
    x2[threadIdx.x] = s;
  }
  __syncthreads();

  const float scl = 0.022097086912079612f;  // TEMP/SCALE/sqrt(128)
  int wave = threadIdx.x >> 5;
  int mb = wave >> 1;
  int nb0 = (wave & 1) << 1;
#pragma unroll
  for (int tt = 0; tt < 2; ++tt) {
    int nb = nb0 + tt;
    v8f acc = {};
#pragma unroll
    for (int kb = 0; kb < 4; ++kb) {
      v16h a  = frag_a_lds(tQ, mb * 16, kb * 32, D);
      v16h bf = frag_b_lds(tQ, nb * 16, kb * 32, D);
      acc = wmma_f16(a, bf, acc);
    }
    int l = laneid();
    int n  = nb * 16 + (l & 15);
    int m0 = mb * 16 + ((l >> 4) << 3);
#pragma unroll
    for (int r = 0; r < 8; ++r) {
      int m = m0 + r;
      S[m * NN + n] = -scl * (x2[m] + x2[n] - 2.f * acc[r]);
    }
  }
  __syncthreads();

  if (threadIdx.x < NN) {
    int t = threadIdx.x;
    float row[NN];
    float mx = -1e30f;
    for (int k = 0; k < NN; ++k) { row[k] = S[t * NN + k]; mx = fmaxf(mx, row[k]); }
    float sum = 0.f;
    for (int k = 0; k < NN; ++k) { row[k] = expf(row[k] - mx); sum += row[k]; }
    float inv = 1.f / sum;
    float amx = -1e30f, amn = 1e30f;
    for (int k = 0; k < NN; ++k) {
      row[k] *= inv;
      amx = fmaxf(amx, row[k]);
      amn = fminf(amn, row[k]);
    }
    float den = 1.f / (amx - amn + 1e-8f);
    long long ab = (long long)w * NN * NN + (long long)t * NN;
    for (int k = 0; k < NN; ++k) af_out[ab + k] = (row[k] - amn) * den;
  }
}

// ---------------------------------------------------------------------------
// k_cluster: greedy clustering per window (64 threads / window)
// ---------------------------------------------------------------------------
__device__ float tril_sum(float* d, float* a) {
  // ascending insertion sort by d, permuting a, then suffix-sum formula
  for (int i = 1; i < NN; ++i) {
    float dk = d[i], ak = a[i];
    int j = i - 1;
    while (j >= 0 && d[j] > dk) { d[j + 1] = d[j]; a[j + 1] = a[j]; --j; }
    d[j + 1] = dk; a[j + 1] = ak;
  }
  float suf = 0.f, res = 0.f;
  for (int j = NN - 1; j >= 0; --j) {
    res += 2.f * d[j] * a[j] * suf;
    suf += a[j];
  }
  return res;
}

__global__ __launch_bounds__(64) void k_cluster(
    const float* __restrict__ af_in, const float* __restrict__ geo,
    float* __restrict__ out_clm, float* __restrict__ out_clgeo,
    float* __restrict__ out_clr, float* __restrict__ out_clcomp,
    float* __restrict__ out_scm, float* __restrict__ clmm_ws) {
  __shared__ float afs[NN * NN];  // clamped af, 16KB
  __shared__ float dens[NN], area[NN], iner[NN], cox[NN], coy[NN];
  __shared__ float compv[NN], logsc[NN];
  __shared__ float clm[CL][NN];
  __shared__ float rowinv[CL];
  __shared__ int selidx;
  int w = blockIdx.x;
  int t = threadIdx.x;
  long long afb = (long long)w * NN * NN;
  for (int i = t; i < NN * NN; i += 64) {
    float v = af_in[afb + i];
    afs[i] = fminf(fmaxf(v, 1e-5f), 0.99999f);  // clamp_pg forward
  }
  {
    long long gb = (long long)w * NN * 6;
    dens[t] = geo[gb + t * 6 + 0];
    area[t] = geo[gb + t * 6 + 2];
    iner[t] = geo[gb + t * 6 + 3];
    cox[t]  = geo[gb + t * 6 + 4];
    coy[t]  = geo[gb + t * 6 + 5];
    logsc[t] = 0.f;
  }
  __syncthreads();

  // initial compactness, row t
  {
    float dl[NN], al[NN];
    float newin = 0.f, sma = 0.f;
    float qx = cox[t], qy = coy[t];
    for (int k = 0; k < NN; ++k) {
      float m_ = afs[t * NN + k];
      float dd = m_ * dens[k];
      float aa = m_ * area[k];
      float mm = dd * aa;
      float dx = qx - cox[k], dy = qy - coy[k];
      newin += dd * iner[k] + mm * (dx * dx + dy * dy);
      sma += mm * aa;
      dl[k] = dd; al[k] = aa;
    }
    float ref = (sma + tril_sum(dl, al)) * (1.f / (2.f * PI_F));
    compv[t] = ref / newin;
  }
  __syncthreads();

  // greedy selection of OUT_CL-1 seeds
  for (int it = 0; it < CL - 1; ++it) {
    if (t == 0) {
      float best = -1e30f;
      int bi = 0;
      for (int k = 0; k < NN; ++k) {
        float sc = compv[k] * expf(logsc[k]);
        if (sc > best) { best = sc; bi = k; }
      }
      selidx = bi;
      out_clr[(long long)w * 7 + it] = (float)bi;
    }
    __syncthreads();
    float rowv = afs[selidx * NN + t];
    clm[it][t] = expf(logsc[t]) * rowv;  // exp(log_sc + log(row))
    logsc[t] += logf(1.f - rowv);
    __syncthreads();
  }
  clm[CL - 1][t] = expf(logsc[t]);  // sc_lm
  __syncthreads();

  // cl_m, sc_m
  for (int i = t; i < CL * NN; i += 64)
    out_clm[(long long)w * CL * NN + i] = clm[i / NN][i % NN];
  out_scm[(long long)w * NN + t] = clm[CL - 1][t];

  // cluster aggregates + compactness (threads 0..7)
  if (t < CL) {
    int c = t;
    float s_cm = 0.f, s_ca = 0.f, sx = 0.f, sy = 0.f;
    for (int k = 0; k < NN; ++k) {
      float m_ = clm[c][k];
      float cd = m_ * dens[k];
      float ca = m_ * area[k];
      float cm = cd * ca;
      s_cm += cm; s_ca += ca;
      sx += cm * cox[k]; sy += cm * coy[k];
    }
    float ccx = sx / (s_cm + 1e-8f), ccy = sy / (s_cm + 1e-8f);
    float cl_mass = s_cm, cl_area = s_ca;
    float cl_dens = cl_mass / (cl_area + 1e-8f);

    float dl[NN], al[NN];
    float newin = 0.f, sma = 0.f;
    for (int k = 0; k < NN; ++k) {
      float m_ = clm[c][k];
      float dd = m_ * dens[k];
      float aa = m_ * area[k];
      float mm = dd * aa;
      float dx = ccx - cox[k], dy = ccy - coy[k];
      newin += dd * iner[k] + mm * (dx * dx + dy * dy);
      sma += mm * aa;
      dl[k] = dd; al[k] = aa;
    }
    float ref = (sma + tril_sum(dl, al)) * (1.f / (2.f * PI_F));
    long long gb = ((long long)w * CL + c) * 6;
    out_clgeo[gb + 0] = cl_dens;
    out_clgeo[gb + 1] = cl_mass;
    out_clgeo[gb + 2] = cl_area;
    out_clgeo[gb + 3] = newin;   // cl_iner
    out_clgeo[gb + 4] = ccx;
    out_clgeo[gb + 5] = ccy;
    out_clcomp[(long long)w * CL + c] = ref / newin;

    rowinv[c] = 1.f / (s_cm * 0.f + [&] {  // rowsum of cl_m
      float s = 0.f;
      for (int k = 0; k < NN; ++k) s += clm[c][k];
      return s;
    }() + 1e-8f);
  }
  __syncthreads();

  // cl_mm = cl_m / (rowsum + 1e-8) -> workspace for k_clf
  for (int i = t; i < CL * NN; i += 64)
    clmm_ws[(long long)w * CL * NN + i] = clm[i / NN][i % NN] * rowinv[i / NN];
}

// ---------------------------------------------------------------------------
// k_clf: cl_f = cl_mm@in_f + cl_mm@f_m + (cl_mm@f_v)@Wfc + bfc  (M padded to 16)
// ---------------------------------------------------------------------------
__global__ __launch_bounds__(256) void k_clf(
    const float* __restrict__ in_f, const float* __restrict__ fm,
    const float* __restrict__ fv, const float* __restrict__ clmm,
    const _Float16* __restrict__ wfch, const float* __restrict__ bfc,
    float* __restrict__ out_clf) {
  __shared__ _Float16 tF[D * NN];  // in_f^T  [n][k] 16KB
  __shared__ _Float16 tM[D * NN];  // f_m^T        16KB
  __shared__ _Float16 tV[D * NN];  // f_v^T        16KB
  __shared__ _Float16 tA[16 * NN]; // cl_mm padded 2KB
  __shared__ _Float16 tT[16 * D];  // E3 (f16)     4KB
  int w = blockIdx.x;
  long long fb = (long long)w * NN * D;
  for (int i = threadIdx.x; i < NN * D; i += 256) {
    int k = i >> 7, n = i & 127;
    tF[n * NN + k] = (_Float16)in_f[fb + i];
    tM[n * NN + k] = (_Float16)fm[fb + i];
    tV[n * NN + k] = (_Float16)fv[fb + i];
  }
  long long cb = (long long)w * CL * NN;
  for (int i = threadIdx.x; i < 16 * NN; i += 256) {
    int r = i >> 6, k = i & 63;
    tA[i] = (r < CL) ? (_Float16)clmm[cb + r * NN + k] : (_Float16)0.f;
  }
  __syncthreads();

  int nb = threadIdx.x >> 5;  // wave id = N block, 0..7
  v8f acc12 = {}, acc3 = {};
#pragma unroll
  for (int kb = 0; kb < 2; ++kb) {
    v16h a = frag_a_lds(tA, 0, kb * 32, NN);
    acc12 = wmma_f16(a, frag_b_lds(tF, nb * 16, kb * 32, NN), acc12);
    acc12 = wmma_f16(a, frag_b_lds(tM, nb * 16, kb * 32, NN), acc12);
    acc3  = wmma_f16(a, frag_b_lds(tV, nb * 16, kb * 32, NN), acc3);
  }
  store_d_lds_h(tT, D, 0, nb * 16, acc3);
  __syncthreads();
#pragma unroll
  for (int kb = 0; kb < 4; ++kb) {
    v16h a = frag_a_lds(tT, 0, kb * 32, D);
    acc12 = wmma_f16(a, frag_b_gl(wfch, kb, nb), acc12);
  }
  int l = laneid();
  if (l < 16) {  // only rows 0..7 are real clusters
    int n = nb * 16 + l;
    float bias = bfc[n];
    long long ob = (long long)w * CL * D;
#pragma unroll
    for (int r = 0; r < 8; ++r) out_clf[ob + r * D + n] = acc12[r] + bias;
  }
}

// ---------------------------------------------------------------------------
extern "C" void kernel_launch(void* const* d_in, const int* in_sizes, int n_in,
                              void* d_out, int out_size, void* d_ws, size_t ws_size,
                              hipStream_t stream) {
  (void)in_sizes; (void)n_in; (void)out_size; (void)ws_size;
  const float* in_f   = (const float*)d_in[0];
  const float* in_geo = (const float*)d_in[1];
  const float* W1  = (const float*)d_in[2];
  const float* b1  = (const float*)d_in[3];
  const float* W2  = (const float*)d_in[4];
  const float* b2  = (const float*)d_in[5];
  const float* Wfc = (const float*)d_in[6];
  const float* bfc = (const float*)d_in[7];
  const float* Wq  = (const float*)d_in[8];
  const float* Wv  = (const float*)d_in[9];
  float* out = (float*)d_out;
  float* ws  = (float*)d_ws;

  float* statF = ws + WS_STATF;
  float* statQ = ws + WS_STATQ;
  _Float16* wh = (_Float16*)(ws + WS_WH);
  _Float16* w1h = wh + 0 * 16384;
  _Float16* w2h = wh + 1 * 16384;
  _Float16* wqh = wh + 2 * 16384;
  _Float16* wvh = wh + 3 * 16384;
  _Float16* wfch = wh + 4 * 16384;
  float* fm   = ws + WS_FM;
  float* fv   = ws + WS_FV;
  float* clmm = ws + WS_CLMM;

  float* out_clf    = out + OFF_CLF;
  float* out_clm    = out + OFF_CLM;
  float* out_clgeo  = out + OFF_CLGEO;
  float* out_clr    = out + OFF_CLR;
  float* out_clcomp = out + OFF_CLCOMP;
  float* out_af     = out + OFF_AF;
  float* out_scm    = out + OFF_SCM;
  float* out_q      = out + OFF_Q;

  const int NWIN = B_ * NW;  // 2048 windows

  k_zero<<<1, 128, 0, stream>>>(ws);
  k_cvtw<<<64, 256, 0, stream>>>(W1, w1h);
  k_cvtw<<<64, 256, 0, stream>>>(W2, w2h);
  k_cvtw<<<64, 256, 0, stream>>>(Wq, wqh);
  k_cvtw<<<64, 256, 0, stream>>>(Wv, wvh);
  k_cvtw<<<64, 256, 0, stream>>>(Wfc, wfch);

  k_stats<<<B_ * 32, 256, 0, stream>>>(in_f, statF);
  k_finstats<<<1, 64, 0, stream>>>(statF);

  k_main<<<NWIN, 256, 0, stream>>>(in_f, statF, w1h, w2h, wqh, wvh, b1, b2,
                                   fm, fv, out_q);

  k_stats<<<B_ * 32, 256, 0, stream>>>(out_q, statQ);
  k_finstats<<<1, 64, 0, stream>>>(statQ);
  k_normq<<<2048, 256, 0, stream>>>(out_q, statQ);

  k_af<<<NWIN, 256, 0, stream>>>(out_q, out_af);
  k_cluster<<<NWIN, 64, 0, stream>>>(out_af, in_geo, out_clm, out_clgeo,
                                     out_clr, out_clcomp, out_scm, clmm);
  k_clf<<<NWIN, 256, 0, stream>>>(in_f, fm, fv, clmm, wfch, bfc, out_clf);
}
